// rnn_crf_47012712022568
// MI455X (gfx1250) — matrix operand order repaired and checked by reading, not compiled, not run on hardware
//
#include <hip/hip_runtime.h>

// ---------------------------------------------------------------------------
// BiLSTM-CRF loss for MI455X (gfx1250), wave32, WMMA bf16 path.
// ---------------------------------------------------------------------------

typedef __attribute__((ext_vector_type(16))) __bf16 v16bf;
typedef __attribute__((ext_vector_type(8)))  float  v8f;

constexpr int kV  = 30000;
constexpr int kDE = 128;
constexpr int kH  = 256;
constexpr int kC  = 16;
constexpr int kB  = 256;
constexpr int kT  = 256;
constexpr int kG  = 4 * kH;          // 1024 gate columns
constexpr int kKT = kDE + kH;        // 384 = fused GEMM K
constexpr int kKTiles  = kKT / 32;   // 12
constexpr int kNTiles  = kG / 16;    // 64
constexpr float kNEG = -10000.0f;
constexpr int SOS = 1, EOS = 2;

__device__ inline __bf16 f2bf(float f) {
  unsigned u = __builtin_bit_cast(unsigned, f);
  unsigned r = u + 0x7fffu + ((u >> 16) & 1u);     // round-to-nearest-even
  unsigned short h = (unsigned short)(r >> 16);
  return __builtin_bit_cast(__bf16, h);
}

// Branch-free fast nonlinearities built on v_exp_f32 / v_rcp_f32 — the libm
// tanhf generated exec-mask divergent polynomial paths on the recurrence's
// critical path.
__device__ inline float fast_sigmoid(float x) {
  return __builtin_amdgcn_rcpf(1.0f + __expf(-x));
}
__device__ inline float fast_tanh(float x) {
  const float cx = fminf(fmaxf(x, -15.0f), 15.0f);  // keep e^{2x} finite
  const float e2 = __expf(2.0f * cx);
  return (e2 - 1.0f) * __builtin_amdgcn_rcpf(e2 + 1.0f);
}

__device__ inline v8f wmma_bf16(v16bf a, v16bf b, v8f c) {
  return __builtin_amdgcn_wmma_f32_16x16x32_bf16(false, a, false, b, (short)0, c,
                                                 false, false);
}

// A fragment: 16x32 bf16 tile starting at column k0 of the LDS A-buffer
// (16 rows x kKT bf16, row-major).  Contiguous k order, as in wprep's B swizzle:
//   lane<16 : M=lane,  K = {k0..k0+15}
//   lane>=16: M=lane-16, K = {k0+16..k0+31}.
__device__ inline v16bf load_a_frag(const __bf16* Abuf, int k0, int lane) {
  const int m    = lane & 15;
  const int half = lane >> 4;
  const __bf16* row = Abuf + m * kKT + k0 + half * 16;
  v16bf a;
#pragma unroll
  for (int e = 0; e < 8; ++e) a[e] = row[e];
#pragma unroll
  for (int e = 0; e < 8; ++e) a[8 + e] = row[8 + e];
  return a;
}

// ---------------------------------------------------------------------------
// Prep 1: gather embeddings -> bf16   xe[b][t][0..127]
// ---------------------------------------------------------------------------
__global__ __launch_bounds__(256) void embed_kernel(const int* __restrict__ x,
                                                    const float* __restrict__ emb,
                                                    __bf16* __restrict__ xe) {
  const size_t tid = (size_t)blockIdx.x * blockDim.x + threadIdx.x; // B*T*16
  const size_t idx = tid >> 4;
  const int    ch  = (int)(tid & 15);
  const int tok = x[idx];
  const float* src = emb + (size_t)tok * kDE + ch * 8;
  __bf16* dst = xe + idx * kDE + ch * 8;
#pragma unroll
  for (int e = 0; e < 8; ++e) dst[e] = f2bf(src[e]);
}

// ---------------------------------------------------------------------------
// Prep 2: convert + swizzle weights into WMMA-fragment-major bf16 layout.
// wsw: [kt][nt][lane(32)][e(16)]  -> each wave's B fragment is contiguous.
//   k = kt*32 + (lane>>4)*16 + e ;  n = nt*16 + (lane&15)
// rows k<128 come from w_ih, else from w_hh.
// wtag (K=256,N=16): 8 k-tiles, single n-tile, same intra-tile swizzle.
// ---------------------------------------------------------------------------
__global__ __launch_bounds__(256) void wprep_kernel(
    const float* __restrict__ w_ih_f, const float* __restrict__ w_hh_f,
    const float* __restrict__ w_ih_b, const float* __restrict__ w_hh_b,
    const float* __restrict__ w_tag,
    __bf16* __restrict__ wsw_f, __bf16* __restrict__ wsw_b,
    __bf16* __restrict__ wtag_top, __bf16* __restrict__ wtag_bot) {
  int i = blockIdx.x * blockDim.x + threadIdx.x;
  const int NW = kKTiles * kNTiles * 512;   // 393216
  if (i < NW) {
    const int kt   = i / (kNTiles * 512);
    const int rem  = i % (kNTiles * 512);
    const int nt   = rem / 512;
    const int q    = rem % 512;
    const int lane = q >> 4;
    const int e    = q & 15;
    const int k = kt * 32 + (lane >> 4) * 16 + e;
    const int n = nt * 16 + (lane & 15);
    const float vf = (k < kDE) ? w_ih_f[k * kG + n] : w_hh_f[(k - kDE) * kG + n];
    const float vb = (k < kDE) ? w_ih_b[k * kG + n] : w_hh_b[(k - kDE) * kG + n];
    wsw_f[i] = f2bf(vf);
    wsw_b[i] = f2bf(vb);
  } else {
    i -= NW;
    if (i < 8 * 512) {
      const int kt   = i / 512;
      const int q    = i % 512;
      const int lane = q >> 4;
      const int e    = q & 15;
      const int k = kt * 32 + (lane >> 4) * 16 + e;  // 0..255
      const int n = lane & 15;
      wtag_top[i] = f2bf(w_tag[(size_t)k * kC + n]);          // rows 0..255
      wtag_bot[i] = f2bf(w_tag[(size_t)(kH + k) * kC + n]);   // rows 256..511
    }
  }
}

// ---------------------------------------------------------------------------
// LSTM direction kernel.  One block owns 16 batch rows for all T steps.
// 512 threads = 16 waves; wave w owns hidden columns [w*16, w*16+16), i.e.
// one 16-col WMMA tile per gate {i,f,g,o}.  Cell state c lives in registers.
// h_t is re-published to LDS as bf16 (A-operand for the next step and for the
// fused tag projection done by wave 0).
//
// NOTE: the K loop is deliberately NOT unrolled — full unroll made the
// compiler hoist 48 B-fragments and spill them to scratch.
// ---------------------------------------------------------------------------
__global__ __launch_bounds__(512) void lstm_kernel(
    const __bf16* __restrict__ xe,      // [B][T][DE] bf16
    const __bf16* __restrict__ wsw,     // swizzled [12][64][512] bf16
    const float*  __restrict__ bias,    // [1024] f32
    const __bf16* __restrict__ wtag_sw, // swizzled [8][512] bf16
    float* __restrict__ score_out,      // [B][T][C] f32 (partial)
    int dir) {
  __shared__ __bf16 Abuf[16 * kKT];     // rows: batch tile; cols: [xe | h]

  const int tid  = threadIdx.x;
  const int lane = tid & 31;
  const int wave = tid >> 5;            // 0..15
  const int m    = lane & 15;
  const int half = lane >> 4;
  const int b0   = blockIdx.x * 16;

  // zero the whole A buffer (h region must start at 0)
  const unsigned short z16 = 0;
  for (int i = tid; i < 16 * kKT; i += blockDim.x)
    Abuf[i] = __builtin_bit_cast(__bf16, z16);

  float c_st[8];
#pragma unroll
  for (int r = 0; r < 8; ++r) c_st[r] = 0.0f;

  // bias for this wave's 4 gate tiles (column = gate*256 + wave*16 + m)
  float bgs[4];
#pragma unroll
  for (int g = 0; g < 4; ++g) bgs[g] = bias[g * kH + wave * 16 + m];

  for (int step = 0; step < kT; ++step) {
    const int t = (dir > 0) ? step : (kT - 1 - step);

    __syncthreads();   // previous step's consumers (incl. wave0 tag GEMM) done

    // stage xe_t -> Abuf[:, 0:128]; 512 threads x 4 bf16 (8B) each
    {
      const int mrow = tid >> 5;        // 0..15
      const int ch   = tid & 31;        // 0..31
      const __bf16* gsrc =
          xe + ((size_t)(b0 + mrow) * kT + t) * kDE + ch * 4;
      *(uint2*)(&Abuf[mrow * kKT + ch * 4]) = *(const uint2*)gsrc;
      // prefetch the next step's row (speculative, L2 temporal)
      const int tn = (dir > 0) ? (t + 1 < kT ? t + 1 : t) : (t > 0 ? t - 1 : t);
      __builtin_prefetch(xe + ((size_t)(b0 + mrow) * kT + tn) * kDE + ch * 4, 0, 1);
    }
    __syncthreads();

    // gates = [xe_t | h_{t-1}] @ Wcat  + bias     (K = 384, bf16 WMMA)
    v8f acc0, acc1, acc2, acc3;
#pragma unroll
    for (int r = 0; r < 8; ++r) {
      acc0[r] = bgs[0]; acc1[r] = bgs[1]; acc2[r] = bgs[2]; acc3[r] = bgs[3];
    }
#pragma unroll 1
    for (int kt = 0; kt < kKTiles; ++kt) {
      const v16bf afrag = load_a_frag(Abuf, kt * 32, lane);
      const __bf16* wb = wsw + ((size_t)kt * kNTiles * 512) + lane * 16;
      const v16bf bf0 = *(const v16bf*)(wb + (size_t)(0 * 16 + wave) * 512);
      const v16bf bf1 = *(const v16bf*)(wb + (size_t)(1 * 16 + wave) * 512);
      const v16bf bf2 = *(const v16bf*)(wb + (size_t)(2 * 16 + wave) * 512);
      const v16bf bf3 = *(const v16bf*)(wb + (size_t)(3 * 16 + wave) * 512);
      acc0 = wmma_bf16(afrag, bf0, acc0);
      acc1 = wmma_bf16(afrag, bf1, acc1);
      acc2 = wmma_bf16(afrag, bf2, acc2);
      acc3 = wmma_bf16(afrag, bf3, acc3);
    }

    // elementwise LSTM cell (all four gates for (row,col) sit in this lane);
    // fully branch-free: v_exp_f32 + v_rcp_f32 only.
    float hreg[8];
#pragma unroll
    for (int r = 0; r < 8; ++r) {
      const float ig = fast_sigmoid(acc0[r]);
      const float fg = fast_sigmoid(acc1[r]);
      const float gg = fast_tanh(acc2[r]);
      const float og = fast_sigmoid(acc3[r]);
      const float c  = fg * c_st[r] + ig * gg;
      c_st[r] = c;
      hreg[r] = og * fast_tanh(c);
    }

    __syncthreads();   // everyone finished reading Abuf for this step

    // publish h_t (bf16) into Abuf[:, 128:384]
#pragma unroll
    for (int r = 0; r < 8; ++r) {
      const int row = r + half * 8;
      Abuf[row * kKT + kDE + wave * 16 + m] = f2bf(hreg[r]);
    }
    __syncthreads();   // h_t visible

    // fused tag projection: scores_part = h_t @ wtag_half  (16x16x256)
    if (wave == 0) {
      v8f s;
#pragma unroll
      for (int r = 0; r < 8; ++r) s[r] = 0.0f;
#pragma unroll 1
      for (int kt = 0; kt < kH / 32; ++kt) {
        const v16bf afrag = load_a_frag(Abuf, kDE + kt * 32, lane);
        const v16bf bfrag = *(const v16bf*)(wtag_sw + (size_t)kt * 512 + lane * 16);
        s = wmma_bf16(afrag, bfrag, s);
      }
#pragma unroll
      for (int r = 0; r < 8; ++r) {
        const int brow = b0 + r + half * 8;
        score_out[((size_t)brow * kT + t) * kC + m] = s[r];
      }
    }
  }
}

// ---------------------------------------------------------------------------
// CRF forward recursion + gold path score; per-batch loss.
// Block = 16 batch rows; 16 threads (one per tag) per batch row.
// ---------------------------------------------------------------------------
__global__ __launch_bounds__(256) void crf_kernel(
    const float* __restrict__ score_f, const float* __restrict__ score_b,
    const float* __restrict__ b_tag, const float* __restrict__ trans,
    const int* __restrict__ x, const int* __restrict__ y,
    float* __restrict__ loss) {
  __shared__ float tr[kC][kC];          // trans[c_new][c_old]
  __shared__ float alpha[16][kC + 1];
  __shared__ float red[16][kC + 1];

  const int tid = threadIdx.x;
  const int bl  = tid >> 4;             // local batch 0..15
  const int c   = tid & 15;             // tag index
  const int b   = blockIdx.x * 16 + bl;

  tr[tid >> 4][tid & 15] = trans[tid];  // 256 threads <-> 256 entries
  const float btag_c = b_tag[c];
  alpha[bl][c] = (c == SOS) ? 0.0f : kNEG;

  float gold = 0.0f;
  int last = SOS, yprev = SOS;
  __syncthreads();

  for (int t = 0; t < kT; ++t) {
    const size_t bt = (size_t)b * kT + t;
    const float mvalid = (x[bt] > 0) ? 1.0f : 0.0f;
    const float sc = (score_f[bt * kC + c] + score_b[bt * kC + c] + btag_c) * mvalid;

    // logsumexp over previous tags
    float mx = kNEG * 4.0f;
#pragma unroll
    for (int k = 0; k < kC; ++k) mx = fmaxf(mx, alpha[bl][k] + tr[c][k]);
    float sum = 0.0f;
#pragma unroll
    for (int k = 0; k < kC; ++k) sum += __expf(alpha[bl][k] + tr[c][k] - mx);
    const float a = mx + __logf(sum) + sc;
    const float newv = (mvalid > 0.0f) ? a : alpha[bl][c];

    const int yt = y[bt];
    if (c == 0) {
      const float emit =
          (score_f[bt * kC + yt] + score_b[bt * kC + yt] + b_tag[yt]) * mvalid;
      gold += (emit + tr[yt][yprev]) * mvalid;
      if (mvalid > 0.0f) last = yt;
    }
    yprev = yt;

    __syncthreads();
    alpha[bl][c] = newv;
    __syncthreads();
  }

  red[bl][c] = alpha[bl][c] + tr[EOS][c];
  __syncthreads();
  if (c == 0) {
    float mx = kNEG * 4.0f;
#pragma unroll
    for (int k = 0; k < kC; ++k) mx = fmaxf(mx, red[bl][k]);
    float sum = 0.0f;
#pragma unroll
    for (int k = 0; k < kC; ++k) sum += __expf(red[bl][k] - mx);
    const float Z = mx + __logf(sum);
    gold += tr[EOS][last];
    loss[b] = Z - gold;
  }
}

__global__ __launch_bounds__(256) void reduce_kernel(const float* __restrict__ loss,
                                                     float* __restrict__ out) {
  __shared__ float sm[256];
  sm[threadIdx.x] = loss[threadIdx.x];
  __syncthreads();
  for (int s = 128; s > 0; s >>= 1) {
    if (threadIdx.x < s) sm[threadIdx.x] += sm[threadIdx.x + s];
    __syncthreads();
  }
  if (threadIdx.x == 0) out[0] = sm[0] * (1.0f / (float)kB);
}

// ---------------------------------------------------------------------------
// Host-side launch
// ---------------------------------------------------------------------------
extern "C" void kernel_launch(void* const* d_in, const int* in_sizes, int n_in,
                              void* d_out, int out_size, void* d_ws, size_t ws_size,
                              hipStream_t stream) {
  const int*   x      = (const int*)d_in[0];
  const int*   y      = (const int*)d_in[1];
  const float* emb    = (const float*)d_in[2];
  const float* w_ih_f = (const float*)d_in[3];
  const float* w_hh_f = (const float*)d_in[4];
  const float* b_f    = (const float*)d_in[5];
  const float* w_ih_b = (const float*)d_in[6];
  const float* w_hh_b = (const float*)d_in[7];
  const float* b_b    = (const float*)d_in[8];
  const float* w_tag  = (const float*)d_in[9];
  const float* b_tag  = (const float*)d_in[10];
  const float* trans  = (const float*)d_in[11];
  float* out = (float*)d_out;

  // workspace carve-up (bytes)
  char* ws = (char*)d_ws;
  size_t off = 0;
  __bf16* xe = (__bf16*)(ws + off);           off += (size_t)kB * kT * kDE * 2;   // 16 MB
  __bf16* wsw_f = (__bf16*)(ws + off);        off += (size_t)kKT * kG * 2;        // 768 KB
  __bf16* wsw_b = (__bf16*)(ws + off);        off += (size_t)kKT * kG * 2;
  __bf16* wtag_top = (__bf16*)(ws + off);     off += (size_t)kH * kC * 2;
  __bf16* wtag_bot = (__bf16*)(ws + off);     off += (size_t)kH * kC * 2;
  float* score_f = (float*)(ws + off);        off += (size_t)kB * kT * kC * 4;    // 4 MB
  float* score_b = (float*)(ws + off);        off += (size_t)kB * kT * kC * 4;
  float* loss_buf = (float*)(ws + off);       off += (size_t)kB * 4;

  // 1) embedding gather -> bf16
  embed_kernel<<<(kB * kT * 16) / 256, 256, 0, stream>>>(x, emb, xe);

  // 2) weight convert + fragment swizzle
  {
    const int total = kKTiles * kNTiles * 512 + 8 * 512;  // 397312
    wprep_kernel<<<(total + 255) / 256, 256, 0, stream>>>(
        w_ih_f, w_hh_f, w_ih_b, w_hh_b, w_tag, wsw_f, wsw_b, wtag_top, wtag_bot);
  }

  // 3) forward and backward LSTM with fused tag projection
  lstm_kernel<<<kB / 16, 512, 0, stream>>>(xe, wsw_f, b_f, wtag_top, score_f, +1);
  lstm_kernel<<<kB / 16, 512, 0, stream>>>(xe, wsw_b, b_b, wtag_bot, score_b, -1);

  // 4) CRF partition function + gold score
  crf_kernel<<<kB / 16, 256, 0, stream>>>(score_f, score_b, b_tag, trans, x, y,
                                          loss_buf);

  // 5) mean over batch
  reduce_kernel<<<1, 256, 0, stream>>>(loss_buf, out);
}
